// SpatioTemporalAttention_71914932404731
// MI455X (gfx1250) — compile-verified, matrix operand order checked by reading, not run on hardware
//
#include <hip/hip_runtime.h>
#include <hip/hip_bf16.h>

typedef __attribute__((ext_vector_type(16))) _Float16 v16h;
typedef __attribute__((ext_vector_type(8)))  _Float16 v8h;
typedef __attribute__((ext_vector_type(8)))  float    v8f;
typedef _Float16 half_t;

#define DEV __device__ __forceinline__

constexpr int Bc = 2, Tc = 3, Hc = 128, Wc = 128, Cc = 128;
constexpr int HEADSc = 8, Dc = 16, CKc = 128;
constexpr int RYc = 2, RXc = 2, PWc = 5, Sc = 25;
constexpr int NPIX = Bc * Tc * Hc * Wc;           // 98304 pixels (rows)

DEV int clampi(int v, int lo, int hi) { return v < lo ? lo : (v > hi ? hi : v); }

// Load one 16-half WMMA fragment (this lane's 16 K-values) from a contiguous
// f16 row: elements 0-7 at [0..7], elements 8-15 at [16..23] (ISA layout with
// the lane-half kb offset already folded into p).  Two b128 loads.
DEV v16h load_frag_h(const half_t* __restrict__ p) {
    v8h lo = *(const v8h*)p;
    v8h hi = *(const v8h*)(p + 16);
    return __builtin_shufflevector(lo, hi, 0, 1, 2, 3, 4, 5, 6, 7,
                                           8, 9, 10, 11, 12, 13, 14, 15);
}

// ---------------------------------------------------------------------------
// Kernel 0: transpose + convert all weight matrices to f16 column-major so
// WMMA B-fragments become two contiguous global_load_b128 per matrix/chunk.
// Wq/Wk/Wv: [C,CK] -> [CK,C].  Wo: [CK,C] -> [C,CK].  128KB total; L2-resident.
// ---------------------------------------------------------------------------
__global__ __launch_bounds__(256)
void cvt_weights_kernel(const float* __restrict__ Wq, const float* __restrict__ Wk,
                        const float* __restrict__ Wv, const float* __restrict__ Wo,
                        half_t* __restrict__ WqT, half_t* __restrict__ WkT,
                        half_t* __restrict__ WvT, half_t* __restrict__ WoT)
{
    int i = blockIdx.x * blockDim.x + threadIdx.x;
    if (i >= Cc * CKc) return;
    int r = i >> 7;          // source row   (K index)
    int c = i & 127;         // source col   (N index)
    WqT[(size_t)c * Cc + r]  = (half_t)Wq[i];
    WkT[(size_t)c * Cc + r]  = (half_t)Wk[i];
    WvT[(size_t)c * Cc + r]  = (half_t)Wv[i];
    WoT[(size_t)c * CKc + r] = (half_t)Wo[i];
}

// ---------------------------------------------------------------------------
// Kernel 1: Q/K/V projections.  One wave computes a 16(pixel) x 32(ck) tile
// (two column tiles) of all three projections: 6 independent f32 accumulators,
// 6 v_wmma_f32_16x16x32_f16 per 32-wide K chunk.  A fragments (values[+temp])
// loaded once per chunk and shared across both column tiles and Q/K.
// ---------------------------------------------------------------------------
__global__ __launch_bounds__(256)
void proj_qkv_kernel(const float* __restrict__ values,
                     const float* __restrict__ temp_emb,
                     const float* __restrict__ sp_emb,
                     const half_t* __restrict__ WqT,
                     const half_t* __restrict__ WkT,
                     const half_t* __restrict__ WvT,
                     half_t* __restrict__ Qh,
                     half_t* __restrict__ Kh,
                     half_t* __restrict__ Vh)
{
    const int lane    = threadIdx.x & 31;
    const int wave    = blockIdx.x * 8 + (threadIdx.x >> 5);
    const int colPair = wave & 3;                 // 2 of the 8 column tiles
    const int rowTile = wave >> 2;
    const int pixBase = rowTile * 16;             // 16 consecutive w positions

    const int wbase = pixBase & (Wc - 1);
    const int h     = (pixBase / Wc) & (Hc - 1);
    const int t     = (pixBase / (Wc * Hc)) % Tc;

    // ISA 16-bit fragment layout: lanes 0-15 -> mn, K 0-7 & 16-23;
    // lanes 16-31 -> mn, K 8-15 & 24-31.
    const int mn   = lane & 15;
    const int kb   = (lane < 16) ? 0 : 8;
    const int col0 = colPair * 32 + mn;
    const int col1 = col0 + 16;

    v8f accQ0 = {}, accQ1 = {}, accK0 = {}, accK1 = {}, accV0 = {}, accV1 = {};

    const float* vrow  = values + (size_t)(pixBase + mn) * Cc;
    const float* terow = temp_emb + t * Cc;

    for (int kc = 0; kc < Cc; kc += 32) {
        // ---- A fragments (row = pixel): f32 contiguous loads + convert ----
        const float* vp = vrow + kc + kb;
        const float* tp = terow + kc + kb;
        v16h aV, aK;
        #pragma unroll
        for (int e = 0; e < 8; ++e) {
            float v0 = vp[e], v1 = vp[16 + e];
            aV[e]     = (half_t)v0;
            aV[8 + e] = (half_t)v1;
            aK[e]     = (half_t)(v0 + tp[e]);
            aK[8 + e] = (half_t)(v1 + tp[16 + e]);
        }
        // ---- B fragments: contiguous f16 rows of transposed weights ----
        v16h bQ0 = load_frag_h(WqT + (size_t)col0 * Cc + kc + kb);
        v16h bQ1 = load_frag_h(WqT + (size_t)col1 * Cc + kc + kb);
        v16h bK0 = load_frag_h(WkT + (size_t)col0 * Cc + kc + kb);
        v16h bK1 = load_frag_h(WkT + (size_t)col1 * Cc + kc + kb);
        v16h bV0 = load_frag_h(WvT + (size_t)col0 * Cc + kc + kb);
        v16h bV1 = load_frag_h(WvT + (size_t)col1 * Cc + kc + kb);

        accQ0 = __builtin_amdgcn_wmma_f32_16x16x32_f16(false, aK, false, bQ0, (short)0, accQ0, false, false);
        accQ1 = __builtin_amdgcn_wmma_f32_16x16x32_f16(false, aK, false, bQ1, (short)0, accQ1, false, false);
        accK0 = __builtin_amdgcn_wmma_f32_16x16x32_f16(false, aK, false, bK0, (short)0, accK0, false, false);
        accK1 = __builtin_amdgcn_wmma_f32_16x16x32_f16(false, aK, false, bK1, (short)0, accK1, false, false);
        accV0 = __builtin_amdgcn_wmma_f32_16x16x32_f16(false, aV, false, bV0, (short)0, accV0, false, false);
        accV1 = __builtin_amdgcn_wmma_f32_16x16x32_f16(false, aV, false, bV1, (short)0, accV1, false, false);
    }

    // ---- store D (lane = column n, VGPR r = row m) ----
    const int mBase = (lane < 16) ? 0 : 8;
    #pragma unroll
    for (int r = 0; r < 8; ++r) {
        int m   = mBase + r;
        int pix = pixBase + m;
        size_t o0 = (size_t)pix * CKc + col0;
        Kh[o0]      = (half_t)accK0[r];
        Kh[o0 + 16] = (half_t)accK1[r];
        Vh[o0]      = (half_t)accV0[r];
        Vh[o0 + 16] = (half_t)accV1[r];
        // query spatial embedding: relative position in clipped patch
        int wq  = wbase + m;
        int cy  = clampi(h,  RYc, Hc - 1 - RYc);
        int cx  = clampi(wq, RXc, Wc - 1 - RXc);
        size_t sb = (size_t)((h - cy + RYc) * PWc + (wq - cx + RXc)) * CKc;
        Qh[o0]      = (half_t)(accQ0[r] + sp_emb[sb + col0]);
        Qh[o0 + 16] = (half_t)(accQ1[r] + sp_emb[sb + col1]);
    }
}

// ---------------------------------------------------------------------------
// Kernel 2: neighborhood attention.  Per-head D=16 dot products with per-query
// shifted 5x5 windows cannot share WMMA operands -> VALU + f16 gathers from
// L2 (K/V f16 = 50MB, fully L2-resident on 192MB).  Two-pass online softmax,
// scores recomputed in pass 2 (no scratch arrays, ~50 VGPRs).
// ---------------------------------------------------------------------------
__global__ __launch_bounds__(256)
void attn_kernel(const half_t* __restrict__ Qh,
                 const half_t* __restrict__ Kh,
                 const half_t* __restrict__ Vh,
                 const float*  __restrict__ sp_emb,
                 half_t* __restrict__ Ah)
{
    int gid = blockIdx.x * blockDim.x + threadIdx.x;
    if (gid >= NPIX * HEADSc) return;
    const int head = gid & (HEADSc - 1);
    const int pix  = gid >> 3;
    const int ww   = pix & (Wc - 1);
    const int hh   = (pix / Wc) & (Hc - 1);
    const int b    = pix / (Wc * Hc * Tc);

    const float scale = 0.25f;  // rsqrt(D=16)

    float q[16];
    const half_t* qp = Qh + (size_t)pix * CKc + head * Dc;
    #pragma unroll
    for (int d = 0; d < 16; ++d) q[d] = (float)qp[d];

    const int cy = clampi(hh, RYc, Hc - 1 - RYc);
    const int cx = clampi(ww, RXc, Wc - 1 - RXc);

    // ---- pass 1: online softmax statistics over 75 keys ----
    float mx = -3.0e38f, denom = 0.0f;
    for (int s = 0; s < Sc; ++s) {
        int ny = clampi(cy + s / PWc - RYc, 0, Hc - 1);
        int nx = clampi(cx + s % PWc - RXc, 0, Wc - 1);
        const float* ep = sp_emb + (size_t)s * CKc + head * Dc;
        float qe = 0.0f;
        #pragma unroll
        for (int d = 0; d < 16; ++d) qe += q[d] * ep[d];
        #pragma unroll
        for (int tt = 0; tt < Tc; ++tt) {
            const half_t* kp = Kh + ((size_t)(((b * Tc + tt) * Hc + ny) * Wc + nx)) * CKc + head * Dc;
            float dot = qe;
            #pragma unroll
            for (int d = 0; d < 16; ++d) dot += q[d] * (float)kp[d];
            float sv = dot * scale;
            float m2 = fmaxf(mx, sv);
            denom = denom * __expf(mx - m2) + __expf(sv - m2);
            mx = m2;
        }
    }
    const float inv = 1.0f / denom;

    // ---- pass 2: recompute scores, accumulate weighted V ----
    float out[16];
    #pragma unroll
    for (int d = 0; d < 16; ++d) out[d] = 0.0f;
    for (int s = 0; s < Sc; ++s) {
        int ny = clampi(cy + s / PWc - RYc, 0, Hc - 1);
        int nx = clampi(cx + s % PWc - RXc, 0, Wc - 1);
        const float* ep = sp_emb + (size_t)s * CKc + head * Dc;
        float qe = 0.0f;
        #pragma unroll
        for (int d = 0; d < 16; ++d) qe += q[d] * ep[d];
        #pragma unroll
        for (int tt = 0; tt < Tc; ++tt) {
            size_t base = ((size_t)(((b * Tc + tt) * Hc + ny) * Wc + nx)) * CKc + head * Dc;
            const half_t* kp = Kh + base;
            float dot = qe;
            #pragma unroll
            for (int d = 0; d < 16; ++d) dot += q[d] * (float)kp[d];
            float wj = __expf(dot * scale - mx) * inv;
            const half_t* vp = Vh + base;
            #pragma unroll
            for (int d = 0; d < 16; ++d) out[d] += wj * (float)vp[d];
        }
    }
    half_t* op = Ah + (size_t)pix * CKc + head * Dc;
    #pragma unroll
    for (int d = 0; d < 16; ++d) op[d] = (half_t)out[d];
}

// ---------------------------------------------------------------------------
// Kernel 3: output projection attn(f16) x Wo -> f32.  All-f16 fragments:
// A and B are both two b128 loads per chunk; 16x32 tile, 2 accumulators.
// ---------------------------------------------------------------------------
__global__ __launch_bounds__(256)
void proj_out_kernel(const half_t* __restrict__ Ah,
                     const half_t* __restrict__ WoT,
                     float* __restrict__ out)
{
    const int lane    = threadIdx.x & 31;
    const int wave    = blockIdx.x * 8 + (threadIdx.x >> 5);
    const int colPair = wave & 3;
    const int rowTile = wave >> 2;
    const int pixBase = rowTile * 16;
    const int mn   = lane & 15;
    const int kb   = (lane < 16) ? 0 : 8;
    const int col0 = colPair * 32 + mn;
    const int col1 = col0 + 16;

    v8f acc0 = {}, acc1 = {};
    const half_t* arow = Ah + (size_t)(pixBase + mn) * CKc;
    for (int kc = 0; kc < CKc; kc += 32) {
        v16h aF = load_frag_h(arow + kc + kb);
        v16h b0 = load_frag_h(WoT + (size_t)col0 * CKc + kc + kb);
        v16h b1 = load_frag_h(WoT + (size_t)col1 * CKc + kc + kb);
        acc0 = __builtin_amdgcn_wmma_f32_16x16x32_f16(false, aF, false, b0, (short)0, acc0, false, false);
        acc1 = __builtin_amdgcn_wmma_f32_16x16x32_f16(false, aF, false, b1, (short)0, acc1, false, false);
    }
    const int mBase = (lane < 16) ? 0 : 8;
    #pragma unroll
    for (int r = 0; r < 8; ++r) {
        size_t o = (size_t)(pixBase + mBase + r) * Cc + col0;
        out[o]      = acc0[r];
        out[o + 16] = acc1[r];
    }
}

// ---------------------------------------------------------------------------
extern "C" void kernel_launch(void* const* d_in, const int* in_sizes, int n_in,
                              void* d_out, int out_size, void* d_ws, size_t ws_size,
                              hipStream_t stream)
{
    const float* values   = (const float*)d_in[0];
    const float* Wq       = (const float*)d_in[1];
    const float* Wk       = (const float*)d_in[2];
    const float* Wv       = (const float*)d_in[3];
    const float* Wo       = (const float*)d_in[4];
    const float* temp_emb = (const float*)d_in[5];
    const float* sp_emb   = (const float*)d_in[6];
    float* out = (float*)d_out;

    const size_t nelem = (size_t)NPIX * CKc;      // 12.58M f16 per buffer
    half_t* Qh  = (half_t*)d_ws;
    half_t* Kh  = Qh + nelem;
    half_t* Vh  = Kh + nelem;
    half_t* Ah  = Vh + nelem;
    half_t* WqT = Ah + nelem;                     // 4 x 16384 f16 weight bufs
    half_t* WkT = WqT + Cc * CKc;
    half_t* WvT = WkT + Cc * CKc;
    half_t* WoT = WvT + Cc * CKc;                 // total ~96.1 MB of d_ws

    cvt_weights_kernel<<<(Cc * CKc + 255) / 256, 256, 0, stream>>>(
        Wq, Wk, Wv, Wo, WqT, WkT, WvT, WoT);

    const int wavesQKV = (NPIX / 16) * 4;         // 16x32 tiles
    proj_qkv_kernel<<<wavesQKV / 8, 256, 0, stream>>>(
        values, temp_emb, sp_emb, WqT, WkT, WvT, Qh, Kh, Vh);

    const int nth = NPIX * HEADSc;                // 786432 query-head threads
    attn_kernel<<<(nth + 255) / 256, 256, 0, stream>>>(Qh, Kh, Vh, sp_emb, Ah);

    proj_out_kernel<<<wavesQKV / 8, 256, 0, stream>>>(Ah, WoT, out);
}
